// SlotAttention_7730941133098
// MI455X (gfx1250) — matrix operand order repaired and checked by reading, not compiled, not run on hardware
//
#include <hip/hip_runtime.h>
#include <hip/hip_bf16.h>

// ---------------------------------------------------------------------------
// MI455X (gfx1250) slot-attention forward.
// GEMMs via v_wmma_f32_16x16x32_bf16 (wave32, 128x128x32 tiles, bf16 in LDS,
// f32 accum, fused bias/relu/residual epilogues). Both A and B tiles are staged
// K-contiguous (pitch 40 shorts) so every fragment is two ds_load_b128s.
// eigh replaced by shift-invert subspace iteration + 16x16 parallel Jacobi.
// ---------------------------------------------------------------------------

typedef __attribute__((ext_vector_type(16))) __bf16 v16bf;
typedef __attribute__((ext_vector_type(8)))  __bf16 v8bf;
typedef __attribute__((ext_vector_type(8)))  float  v8f;

#define NB 64
#define NTOK 256
#define ND 512
#define NS 8
#define MAXS 64
#define SCALE_QK 0.04419417382415922f   /* 512^-0.5 */

__device__ __forceinline__ float blockReduceSum(float v, float* red) {
  int t = threadIdx.x;
  red[t] = v; __syncthreads();
  #pragma unroll
  for (int s = 128; s > 0; s >>= 1) {
    if (t < s) red[t] += red[t + s];
    __syncthreads();
  }
  float r = red[0]; __syncthreads();
  return r;
}

// ------------------------------ position embeds -----------------------------
__global__ void encpos_kernel(const float* __restrict__ w, const float* __restrict__ b,
                              float* __restrict__ pos, float* __restrict__ post) {
  int g = blockIdx.x * 256 + threadIdx.x;          // < 256*512
  int n = g >> 9, d = g & 511;
  int ri = n >> 4, rj = n & 15;
  float gx = ri * (1.0f / 15.0f), gy = rj * (1.0f / 15.0f);
  float v = gx * w[d] + gy * w[512 + d] + (1.0f - gx) * w[1024 + d]
          + (1.0f - gy) * w[1536 + d] + b[d];
  pos[n * 512 + d] = v;       // (n,d)
  post[d * 256 + n] = v;      // (d,n) for coalesced stats pass
}

__global__ void decpos_kernel(const float* __restrict__ w, const float* __restrict__ b,
                              float* __restrict__ dpos) {
  int g = blockIdx.x * 256 + threadIdx.x;          // < 512*256
  int d = g >> 8, r = g & 255;
  int ri = r >> 4, rj = r & 15;
  float gx = ri * (1.0f / 15.0f), gy = rj * (1.0f / 15.0f);
  dpos[d * 256 + r] = gx * w[d] + gy * w[512 + d] + (1.0f - gx) * w[1024 + d]
                    + (1.0f - gy) * w[1536 + d] + b[d];
}

// -------------------- per-batch LN((N,D)) stats & apply ----------------------
__global__ void stats_kernel(const float* __restrict__ x, const float* __restrict__ post,
                             float* __restrict__ stats) {
  __shared__ float red[256];
  int bb = blockIdx.x;
  const float* xb = x + (long long)bb * (ND * NTOK);
  int t = threadIdx.x;
  float s1 = 0, s2 = 0, s3 = 0, s4 = 0;
  for (int i = t; i < ND * NTOK; i += 256) {
    float e = xb[i];
    float f = e + post[i];
    s1 += e; s2 += e * e; s3 += f; s4 += f * f;
  }
  s1 = blockReduceSum(s1, red);
  s2 = blockReduceSum(s2, red);
  s3 = blockReduceSum(s3, red);
  s4 = blockReduceSum(s4, red);
  if (t == 0) {
    float inv = 1.0f / (float)(ND * NTOK);
    float m1 = s1 * inv, m2 = s3 * inv;
    stats[bb * 4 + 0] = m1;
    stats[bb * 4 + 1] = rsqrtf(s2 * inv - m1 * m1 + 1e-5f);
    stats[bb * 4 + 2] = m2;
    stats[bb * 4 + 3] = rsqrtf(s4 * inv - m2 * m2 + 1e-5f);
  }
}

// x (b,d,n) -> transpose LN -> f_np (b,n,d) and fp_ln (b,n,d) in one pass.
__global__ void lnapply_kernel(const float* __restrict__ x, const float* __restrict__ pos,
                               const float* __restrict__ stats,
                               const float* __restrict__ g, const float* __restrict__ bb,
                               float* __restrict__ fnp, float* __restrict__ fpln) {
  __shared__ float tile[32][33];
  int b = blockIdx.z;
  int d0 = blockIdx.x * 32, n0 = blockIdx.y * 32;
  int tx = threadIdx.x, ty = threadIdx.y;          // (32,8)
  const float* xb = x + (long long)b * (ND * NTOK);
  #pragma unroll
  for (int p = 0; p < 4; p++) {
    int dl = ty + p * 8;
    tile[dl][tx] = xb[(d0 + dl) * 256 + n0 + tx];  // coalesced over n
  }
  __syncthreads();
  float mf = stats[b * 4 + 0], rf = stats[b * 4 + 1];
  float mp = stats[b * 4 + 2], rp = stats[b * 4 + 3];
  #pragma unroll
  for (int p = 0; p < 4; p++) {
    int nl = ty + p * 8;
    int n = n0 + nl, d = d0 + tx;
    float v = tile[tx][nl];
    int gi = n * ND + d;
    long long go = (long long)b * (NTOK * ND) + gi;
    float gg = g[gi], bv = bb[gi];
    fnp[go]  = (v - mf) * rf * gg + bv;
    fpln[go] = (v + pos[gi] - mp) * rp * gg + bv;
  }
}

// ------------------------------- row LN / norm -------------------------------
__global__ void rowln_kernel(const float* __restrict__ in, float* __restrict__ out,
                             const float* __restrict__ g, const float* __restrict__ b, int D) {
  __shared__ float red[256];
  long long row = blockIdx.x;
  const float* x = in + row * D;
  float* y = out + row * D;
  int t = threadIdx.x;
  float s = 0, s2 = 0;
  for (int d = t; d < D; d += 256) { float v = x[d]; s += v; s2 += v * v; }
  s  = blockReduceSum(s, red);
  s2 = blockReduceSum(s2, red);
  float mean = s / D;
  float rinv = rsqrtf(s2 / D - mean * mean + 1e-5f);
  for (int d = t; d < D; d += 256) y[d] = (x[d] - mean) * rinv * g[d] + b[d];
}

__global__ void rownorm_kernel(float* __restrict__ x, int D) {
  __shared__ float red[256];
  long long row = blockIdx.x;
  float* p = x + row * D;
  int t = threadIdx.x;
  float s = 0;
  for (int d = t; d < D; d += 256) { float v = p[d]; s += v * v; }
  s = blockReduceSum(s, red);
  float inv = 1.0f / fmaxf(sqrtf(s), 1e-12f);
  for (int d = t; d < D; d += 256) p[d] *= inv;
}

// ------------------------------ WMMA bf16 GEMM -------------------------------
// C[M,N] = (relu?)(A[M,K] @ B + bias) (+res). B is (K,N) or (N,K) if transB.
// 128x128x32 tiles, 8 waves (2x4), each wave 4x2 WMMA 16x16x32 bf16 subtiles.
// LDS tiles: lA[row][k] and lBt[col][k], pitch 40 shorts -> every fragment is
// two 16B-aligned ds_load_b128s.
__launch_bounds__(256, 2)
__global__ void gemm_wmma_bf16(const float* __restrict__ A, const float* __restrict__ Bw,
                               const float* __restrict__ bias, const float* __restrict__ res,
                               float* __restrict__ C, int M, int N, int K,
                               long long sA, long long sB, long long sC,
                               int transB, int doRelu) {
  __shared__ __bf16 lA[128 * 40];
  __shared__ __bf16 lBt[128 * 40];   // B tile stored transposed: [n][k]
  const int tid = threadIdx.x;
  const int zb = blockIdx.z;
  A  += (long long)zb * sA;
  Bw += (long long)zb * sB;
  C  += (long long)zb * sC;
  const float* resz = res ? res + (long long)zb * sC : nullptr;
  const int m0 = blockIdx.y * 128;
  const int n0 = blockIdx.x * 128;
  const int lane = tid & 31, wid = tid >> 5;
  const int wm = wid >> 2, wn = wid & 3;           // 2x4 wave grid
  const int half = lane >> 4, l16 = lane & 15;

  v8f acc[4][2];
  #pragma unroll
  for (int mt = 0; mt < 4; mt++)
    #pragma unroll
    for (int nt = 0; nt < 2; nt++)
      #pragma unroll
      for (int e = 0; e < 8; e++) acc[mt][nt][e] = 0.0f;

  for (int k0 = 0; k0 < K; k0 += 32) {
    { // A tile 128x32, f32 -> bf16, K-contiguous rows
      int c4 = (tid & 7) * 4, rb = tid >> 3;
      #pragma unroll
      for (int p = 0; p < 4; p++) {
        int r = rb + p * 32, gr = m0 + r;
        float4 v4 = make_float4(0.f, 0.f, 0.f, 0.f);
        if (gr < M) v4 = *(const float4*)(A + (long long)gr * K + k0 + c4);
        lA[r * 40 + c4 + 0] = (__bf16)v4.x;
        lA[r * 40 + c4 + 1] = (__bf16)v4.y;
        lA[r * 40 + c4 + 2] = (__bf16)v4.z;
        lA[r * 40 + c4 + 3] = (__bf16)v4.w;
        if (gr < M && k0 + 32 < K)
          __builtin_prefetch(A + (long long)gr * K + k0 + 32 + c4, 0, 1);
      }
    }
    if (!transB) {
      // B (K,N): 4 coalesced column reads per thread, contiguous LDS writes.
      int nn = tid & 127;            // column within tile
      int kb = (tid >> 7) * 4;       // 0 or 4
      #pragma unroll
      for (int p = 0; p < 4; p++) {
        int kk = kb + p * 8;
        const float* src = Bw + (long long)(k0 + kk) * N + n0 + nn;
        float f0 = src[0];
        float f1 = src[(long long)N];
        float f2 = src[(long long)N * 2];
        float f3 = src[(long long)N * 3];
        lBt[nn * 40 + kk + 0] = (__bf16)f0;
        lBt[nn * 40 + kk + 1] = (__bf16)f1;
        lBt[nn * 40 + kk + 2] = (__bf16)f2;
        lBt[nn * 40 + kk + 3] = (__bf16)f3;
      }
      if (k0 + 32 < K) __builtin_prefetch(Bw + (long long)(k0 + 32) * N + n0 + nn, 0, 1);
    } else {
      // B (N,K): float4 along K, contiguous LDS writes.
      int k4 = (tid & 7) * 4, nb2 = tid >> 3;
      #pragma unroll
      for (int p = 0; p < 4; p++) {
        int nn = nb2 + p * 32;
        float4 v4 = *(const float4*)(Bw + (long long)(n0 + nn) * K + k0 + k4);
        lBt[nn * 40 + k4 + 0] = (__bf16)v4.x;
        lBt[nn * 40 + k4 + 1] = (__bf16)v4.y;
        lBt[nn * 40 + k4 + 2] = (__bf16)v4.z;
        lBt[nn * 40 + k4 + 3] = (__bf16)v4.w;
        if (k0 + 32 < K)
          __builtin_prefetch(Bw + (long long)(n0 + nn) * K + k0 + 32 + k4, 0, 1);
      }
    }
    __syncthreads();

    // A fragment: elements 0..7 -> K = half*8 + i, 8..15 -> K = 16 + half*8 + i
    v16bf af[4];
    #pragma unroll
    for (int mt = 0; mt < 4; mt++) {
      int row = wm * 64 + mt * 16 + l16;
      const __bf16* ap = &lA[row * 40 + half * 8];
      v8bf a0 = *(const v8bf*)(ap);        // 16B aligned (row*80 + half*16)
      v8bf a1 = *(const v8bf*)(ap + 16);   // +32B
      af[mt] = __builtin_shufflevector(a0, a1, 0, 1, 2, 3, 4, 5, 6, 7,
                                       8, 9, 10, 11, 12, 13, 14, 15);
    }
    // B fragment: elements 0..7 -> K = half*8 + i, 8..15 -> K = 16 + half*8 + i
    v16bf bfr[2];
    #pragma unroll
    for (int nt = 0; nt < 2; nt++) {
      int col = wn * 32 + nt * 16 + l16;
      const __bf16* bp = &lBt[col * 40 + half * 8];
      v8bf b0 = *(const v8bf*)(bp);        // 16B aligned (col*80 + half*16)
      v8bf b1 = *(const v8bf*)(bp + 16);   // +32B
      bfr[nt] = __builtin_shufflevector(b0, b1, 0, 1, 2, 3, 4, 5, 6, 7,
                                        8, 9, 10, 11, 12, 13, 14, 15);
    }
    #pragma unroll
    for (int mt = 0; mt < 4; mt++)
      #pragma unroll
      for (int nt = 0; nt < 2; nt++)
        acc[mt][nt] = __builtin_amdgcn_wmma_f32_16x16x32_bf16(
            false, af[mt], false, bfr[nt], (short)0, acc[mt][nt], false, false);
    __syncthreads();
  }

  // epilogue: D element r -> row M = half*8 + r, col = l16
  #pragma unroll
  for (int mt = 0; mt < 4; mt++) {
    #pragma unroll
    for (int nt = 0; nt < 2; nt++) {
      int col = n0 + wn * 32 + nt * 16 + l16;
      float bv = bias ? bias[col] : 0.0f;
      #pragma unroll
      for (int r = 0; r < 8; r++) {
        int gr = m0 + wm * 64 + mt * 16 + half * 8 + r;
        if (gr < M) {
          float v = acc[mt][nt][r] + bv;
          if (doRelu) v = fmaxf(v, 0.0f);
          long long o = (long long)gr * N + col;
          if (resz) v += resz[o];
          C[o] = v;
        }
      }
    }
  }
}

// ---------------- shift-invert subspace iteration (eigh replacement) --------
__global__ void subspace_kernel(const float* __restrict__ cov, float* __restrict__ Q, int first) {
  __shared__ float sQ[256 * 16];
  __shared__ float sY[256 * 16];
  __shared__ float red[256];
  int b = blockIdx.x;
  const float* cb = cov + (long long)b * 65536;
  float* qb = Q + (long long)b * 4096;
  int t = threadIdx.x;
  if (first) {
    #pragma unroll
    for (int c = 0; c < 16; c++) {
      unsigned h = (unsigned)(b * 4096 + t * 16 + c) * 2654435761u;
      h ^= h >> 13; h *= 0x5bd1e995u; h ^= h >> 15;
      sQ[t * 16 + c] = ((float)(h & 0xFFFFu)) * (1.0f / 65536.0f) - 0.5f;
    }
  } else {
    #pragma unroll
    for (int c = 0; c < 16; c++) sQ[t * 16 + c] = qb[t * 16 + c];
  }
  __syncthreads();
  float acc[16];
  #pragma unroll
  for (int c = 0; c < 16; c++) acc[c] = 260.0f * sQ[t * 16 + c];  // shift > max eig (trace=256)
  for (int k = 0; k < 256; k++) {
    float cv = cb[t * 256 + k];
    #pragma unroll
    for (int c = 0; c < 16; c++) acc[c] -= cv * sQ[k * 16 + c];
  }
  #pragma unroll
  for (int c = 0; c < 16; c++) sY[t * 16 + c] = acc[c];
  __syncthreads();
  // modified Gram-Schmidt
  for (int c = 0; c < 16; c++) {
    for (int p = 0; p < c; p++) {
      float d = blockReduceSum(sY[t * 16 + p] * sY[t * 16 + c], red);
      sY[t * 16 + c] -= d * sY[t * 16 + p];
    }
    float n2 = blockReduceSum(sY[t * 16 + c] * sY[t * 16 + c], red);
    sY[t * 16 + c] *= rsqrtf(fmaxf(n2, 1e-20f));
    __syncthreads();
  }
  #pragma unroll
  for (int c = 0; c < 16; c++) qb[t * 16 + c] = sY[t * 16 + c];
}

// Rayleigh-Ritz: T = Q^T cov Q, parallel 16x16 Jacobi, ascending ranks 1..8.
__global__ void rayleigh_kernel(const float* __restrict__ cov, const float* __restrict__ Q,
                                float* __restrict__ eraw) {
  __shared__ float sQ[256 * 16];
  __shared__ float sZ[256 * 16];
  __shared__ float sT[256];
  __shared__ float sU[256];
  __shared__ int sperm[16];
  int b = blockIdx.x;
  const float* cb = cov + (long long)b * 65536;
  const float* qb = Q + (long long)b * 4096;
  int t = threadIdx.x;
  #pragma unroll
  for (int c = 0; c < 16; c++) sQ[t * 16 + c] = qb[t * 16 + c];
  __syncthreads();
  float acc[16];
  #pragma unroll
  for (int c = 0; c < 16; c++) acc[c] = 0.0f;
  for (int k = 0; k < 256; k++) {
    float cv = cb[t * 256 + k];
    #pragma unroll
    for (int c = 0; c < 16; c++) acc[c] += cv * sQ[k * 16 + c];
  }
  #pragma unroll
  for (int c = 0; c < 16; c++) sZ[t * 16 + c] = acc[c];
  __syncthreads();
  { // T = Q^T Z, one (i,j) per thread
    int i = t >> 4, j = t & 15;
    float s = 0;
    for (int n = 0; n < 256; n++) s += sQ[n * 16 + i] * sZ[n * 16 + j];
    sT[i * 16 + j] = s;
    sU[t] = (i == j) ? 1.0f : 0.0f;
  }
  __syncthreads();
  int arr[16];
  #pragma unroll
  for (int i = 0; i < 16; i++) arr[i] = i;
  for (int sweep = 0; sweep < 8; sweep++) {
    for (int rnd = 0; rnd < 15; rnd++) {
      float cth = 1.0f, sth = 0.0f;
      int p = 0, q = 1;
      if (t < 8) {
        p = arr[t]; q = arr[15 - t];
        if (p > q) { int tmp = p; p = q; q = tmp; }
        float app = sT[p * 16 + p], aqq = sT[q * 16 + q], apq = sT[p * 16 + q];
        if (fabsf(apq) > 1e-20f) {
          float tau = (aqq - app) / (2.0f * apq);
          float tt = (tau >= 0.0f ? 1.0f : -1.0f) / (fabsf(tau) + sqrtf(1.0f + tau * tau));
          cth = rsqrtf(1.0f + tt * tt);
          sth = tt * cth;
        }
      }
      __syncthreads();
      if (t < 8) {  // row phase (disjoint rows per pair)
        for (int k = 0; k < 16; k++) {
          float a = sT[p * 16 + k], c2 = sT[q * 16 + k];
          sT[p * 16 + k] = cth * a - sth * c2;
          sT[q * 16 + k] = sth * a + cth * c2;
        }
      }
      __syncthreads();
      if (t < 8) {  // column phase + eigenvector accumulation
        for (int k = 0; k < 16; k++) {
          float a = sT[k * 16 + p], c2 = sT[k * 16 + q];
          sT[k * 16 + p] = cth * a - sth * c2;
          sT[k * 16 + q] = sth * a + cth * c2;
          float u = sU[k * 16 + p], u2 = sU[k * 16 + q];
          sU[k * 16 + p] = cth * u - sth * u2;
          sU[k * 16 + q] = sth * u + cth * u2;
        }
      }
      __syncthreads();
      int last = arr[15];
      for (int i = 15; i >= 2; i--) arr[i] = arr[i - 1];
      arr[1] = last;
    }
  }
  if (t == 0) {  // stable ascending sort of Ritz values
    float vals[16]; int ord[16];
    for (int i = 0; i < 16; i++) { vals[i] = sT[i * 16 + i]; ord[i] = i; }
    for (int i = 1; i < 16; i++) {
      float v = vals[i]; int o = ord[i]; int j = i - 1;
      while (j >= 0 && vals[j] > v) { vals[j + 1] = vals[j]; ord[j + 1] = ord[j]; j--; }
      vals[j + 1] = v; ord[j + 1] = o;
    }
    for (int i = 0; i < 16; i++) sperm[i] = ord[i];
  }
  __syncthreads();
  for (int s = 0; s < NS; s++) {   // skip rank 0 (evec[:, :, 1:NS+1])
    int j = sperm[s + 1];
    float v = 0;
    #pragma unroll
    for (int c = 0; c < 16; c++) v += sQ[t * 16 + c] * sU[c * 16 + j];
    eraw[((long long)b * NS + s) * NTOK + t] = v;
  }
}

// ------------------------------ quantization ---------------------------------
__global__ void quant_kernel(const float* __restrict__ z, const float* __restrict__ tsl,
                             float* __restrict__ dist, int* __restrict__ idx0) {
  __shared__ float rd[64];
  __shared__ int ri[64];
  int row = blockIdx.x;                // b*NS+s
  int t = threadIdx.x;                 // candidate slot 0..63
  const float* zr = z + (long long)row * ND;
  const float* tr = tsl + (long long)t * ND;
  float s = 0;
  for (int d = 0; d < ND; d++) { float df = zr[d] - tr[d]; s += df * df; }
  rd[t] = s; ri[t] = t;
  __syncthreads();
  for (int st = 32; st > 0; st >>= 1) {
    if (t < st) {
      if (rd[t + st] < rd[t] || (rd[t + st] == rd[t] && ri[t + st] < ri[t])) {
        rd[t] = rd[t + st]; ri[t] = ri[t + st];
      }
    }
    __syncthreads();
  }
  if (t == 0) { dist[row] = rd[0]; idx0[row] = ri[0]; }
}

// per-batch argsort of 8 dists; faithful cross-batch quirk: idx[order] with
// order values 0..7 used as GLOBAL indices into flat idx.
__global__ void order_kernel(const float* __restrict__ dist, const int* __restrict__ idx0,
                             int* __restrict__ fidx, float* __restrict__ outIdx) {
  int b = threadIdx.x;
  if (b < NB) {
    float v[8]; int o[8];
    for (int s = 0; s < 8; s++) { v[s] = dist[b * 8 + s]; o[s] = s; }
    for (int i = 1; i < 8; i++) {
      float vv = v[i]; int oo = o[i]; int j = i - 1;
      while (j >= 0 && v[j] > vv) { v[j + 1] = v[j]; o[j + 1] = o[j]; j--; }
      v[j + 1] = vv; o[j + 1] = oo;
    }
    for (int s = 0; s < 8; s++) {
      int id = idx0[o[s]];             // global (first-batch) gather — quirk
      fidx[b * 8 + s] = id;
      outIdx[b * 8 + s] = (float)id;
    }
  }
}

__global__ void slotsinit_kernel(const float* __restrict__ emb, const float* __restrict__ tsl,
                                 const float* __restrict__ z, const int* __restrict__ fidx,
                                 float* __restrict__ slots, float* __restrict__ qpart) {
  __shared__ float red[256];
  int g = blockIdx.x * 256 + threadIdx.x;   // < B*NS*D
  int bs = g >> 9, d = g & 511;
  int id = fidx[bs];
  float zq = emb[id * ND + d];
  float tz = tsl[id * ND + d];
  float df = tz - z[g];
  slots[g] = zq;                            // straight-through forward value
  float s = blockReduceSum(df * df, red);
  if (threadIdx.x == 0) qpart[blockIdx.x] = s;
}

__global__ void qloss_kernel(const float* __restrict__ qpart, float* __restrict__ out) {
  __shared__ float red[256];
  int t = threadIdx.x;
  float s = qpart[t] + qpart[t + 256] + qpart[t + 512] + qpart[t + 768];
  s = blockReduceSum(s, red);
  if (t == 0) out[0] = s * (1.99f / 64.0f);   // (1+BETA)/B
}

// ----------------------------- slot attention --------------------------------
__global__ void attn_kernel(const float* __restrict__ q, const float* __restrict__ k,
                            const float* __restrict__ v, float* __restrict__ upd) {
  __shared__ float sq[NS * ND];
  __shared__ float sa[NS * NTOK];
  __shared__ float red[256];
  int b = blockIdx.x;
  int t = threadIdx.x;
  for (int i = t; i < NS * ND; i += 256) sq[i] = q[(long long)b * NS * ND + i];
  __syncthreads();
  const float* kr = k + ((long long)b * NTOK + t) * ND;
  float dots[NS];
  #pragma unroll
  for (int i = 0; i < NS; i++) dots[i] = 0.0f;
  for (int d = 0; d < ND; d++) {
    float kv = kr[d];
    #pragma unroll
    for (int i = 0; i < NS; i++) dots[i] += sq[i * ND + d] * kv;
  }
  float mx = -1e30f;
  #pragma unroll
  for (int i = 0; i < NS; i++) { dots[i] *= SCALE_QK; mx = fmaxf(mx, dots[i]); }
  float se = 0;
  #pragma unroll
  for (int i = 0; i < NS; i++) { float e = expf(dots[i] - mx); dots[i] = e; se += e; }
  float inv = 1.0f / se;
  #pragma unroll
  for (int i = 0; i < NS; i++) sa[i * NTOK + t] = dots[i] * inv + 1e-8f;  // softmax over slots + EPS
  __syncthreads();
  for (int i = 0; i < NS; i++) {        // renormalize over tokens
    float rs = blockReduceSum(sa[i * NTOK + t], red);
    sa[i * NTOK + t] /= rs;
    __syncthreads();
  }
  for (int dd = 0; dd < 2; dd++) {      // updates = attn @ v
    int d = t + dd * 256;
    float acc2[NS];
    #pragma unroll
    for (int i = 0; i < NS; i++) acc2[i] = 0.0f;
    for (int j = 0; j < NTOK; j++) {
      float vv = v[((long long)b * NTOK + j) * ND + d];
      #pragma unroll
      for (int i = 0; i < NS; i++) acc2[i] += sa[i * NTOK + j] * vv;
    }
    #pragma unroll
    for (int i = 0; i < NS; i++) upd[((long long)b * NS + i) * ND + d] = acc2[i];
  }
}

__global__ void gru_kernel(const float* __restrict__ gi, const float* __restrict__ gh,
                           float* __restrict__ slots) {
  long long g = (long long)blockIdx.x * 256 + threadIdx.x;   // < B*NS*D
  int row = (int)(g >> 9), d = (int)(g & 511);
  long long base = (long long)row * 1536;
  float ir = gi[base + d], iz = gi[base + 512 + d], in = gi[base + 1024 + d];
  float hr = gh[base + d], hz = gh[base + 512 + d], hn = gh[base + 1024 + d];
  float r  = 1.0f / (1.0f + expf(-(ir + hr)));
  float zz = 1.0f / (1.0f + expf(-(iz + hz)));
  float n  = tanhf(in + r * hn);
  float h  = slots[g];
  slots[g] = (1.0f - zz) * n + zz * h;
}

// ------------------------------- decoder / IO --------------------------------
__global__ void decoder_kernel(const float* __restrict__ slots, const float* __restrict__ dpos,
                               float* __restrict__ out) {
  long long g = (long long)blockIdx.x * 256 + threadIdx.x;
  int r = (int)(g & 255);
  int d = (int)((g >> 8) & 511);
  long long bs = g >> 17;
  out[g] = slots[bs * ND + d] + dpos[d * 256 + r];
}

__global__ void copyslots_kernel(const float* __restrict__ slots, float* __restrict__ out) {
  int g = blockIdx.x * 256 + threadIdx.x;
  out[g] = slots[g];
}

// --------------------------------- driver ------------------------------------
extern "C" void kernel_launch(void* const* d_in, const int* in_sizes, int n_in,
                              void* d_out, int out_size, void* d_ws, size_t ws_size,
                              hipStream_t stream) {
  (void)in_sizes; (void)n_in; (void)out_size; (void)ws_size;
  const float* x          = (const float*)d_in[0];
  const float* enc_pos_w  = (const float*)d_in[1];
  const float* enc_pos_b  = (const float*)d_in[2];
  const float* enc_norm_g = (const float*)d_in[3];
  const float* enc_norm_b = (const float*)d_in[4];
  const float* enc_mlp_w1 = (const float*)d_in[5];
  const float* enc_mlp_b1 = (const float*)d_in[6];
  const float* enc_mlp_w2 = (const float*)d_in[7];
  const float* enc_mlp_b2 = (const float*)d_in[8];
  const float* anch_w1    = (const float*)d_in[9];
  const float* anch_b1    = (const float*)d_in[10];
  const float* anch_w2    = (const float*)d_in[11];
  const float* anch_b2    = (const float*)d_in[12];
  const float* wq         = (const float*)d_in[13];
  const float* bq         = (const float*)d_in[14];
  const float* wk         = (const float*)d_in[15];
  const float* bk         = (const float*)d_in[16];
  const float* wv         = (const float*)d_in[17];
  const float* bv         = (const float*)d_in[18];
  const float* gru_wih    = (const float*)d_in[19];
  const float* gru_whh    = (const float*)d_in[20];
  const float* gru_bih    = (const float*)d_in[21];
  const float* gru_bhh    = (const float*)d_in[22];
  const float* mlp_w1     = (const float*)d_in[23];
  const float* mlp_b1     = (const float*)d_in[24];
  const float* mlp_w2     = (const float*)d_in[25];
  const float* mlp_b2     = (const float*)d_in[26];
  const float* ni_g       = (const float*)d_in[27];
  const float* ni_b       = (const float*)d_in[28];
  const float* ns_g       = (const float*)d_in[29];
  const float* ns_b       = (const float*)d_in[30];
  const float* npf_g      = (const float*)d_in[31];
  const float* npf_b      = (const float*)d_in[32];
  const float* emb        = (const float*)d_in[33];
  const float* dec_pos_w  = (const float*)d_in[34];
  const float* dec_pos_b  = (const float*)d_in[35];

  // workspace layout (floats)
  constexpr long long BIGS = (long long)NB * NTOK * ND;           // 8388608
  constexpr long long OFF_A     = 0;                 // f_np
  constexpr long long OFF_B     = OFF_A + BIGS;      // fp_ln -> fp
  constexpr long long OFF_C     = OFF_B + BIGS;      // t1 -> k
  constexpr long long OFF_D     = OFF_C + BIGS;      // k_nopos -> xn -> v
  constexpr long long OFF_COV   = OFF_D + BIGS;      // 4194304 (reused: gi/gh)
  constexpr long long OFF_POS   = OFF_COV + (long long)NB * NTOK * NTOK;
  constexpr long long OFF_POST  = OFF_POS + NTOK * ND;
  constexpr long long OFF_STATS = OFF_POST + ND * NTOK;
  constexpr long long OFF_Q     = OFF_STATS + 256;
  constexpr long long OFF_ERAW  = OFF_Q + (long long)NB * NTOK * 16;
  constexpr long long OFF_E1    = OFF_ERAW + (long long)NB * NS * NTOK;
  constexpr long long OFF_EVEC  = OFF_E1 + (long long)NB * NS * NTOK;
  constexpr long long OFF_Z     = OFF_EVEC + (long long)NB * NS * NTOK;
  constexpr long long OFF_TSL   = OFF_Z + (long long)NB * NS * ND;
  constexpr long long OFF_DIST  = OFF_TSL + (long long)MAXS * ND;
  constexpr long long OFF_IDX0  = OFF_DIST + 512;
  constexpr long long OFF_FIDX  = OFF_IDX0 + 512;
  constexpr long long OFF_SLOTS = OFF_FIDX + 512;
  constexpr long long OFF_LNB   = OFF_SLOTS + (long long)NB * NS * ND;
  constexpr long long OFF_QB    = OFF_LNB + (long long)NB * NS * ND;
  constexpr long long OFF_UPD   = OFF_QB + (long long)NB * NS * ND;
  constexpr long long OFF_H1    = OFF_UPD + (long long)NB * NS * ND;
  constexpr long long OFF_DPOS  = OFF_H1 + (long long)NB * NS * ND;
  constexpr long long OFF_QP    = OFF_DPOS + (long long)ND * NTOK;
  constexpr long long OFF_GI    = OFF_COV;                 // cov dead by then
  constexpr long long OFF_GH    = OFF_COV + (long long)512 * 1536;

  float* ws = (float*)d_ws;
  float* FNP  = ws + OFF_A;
  float* FB   = ws + OFF_B;
  float* KC   = ws + OFF_C;
  float* DV   = ws + OFF_D;
  float* COV  = ws + OFF_COV;
  float* POS  = ws + OFF_POS;
  float* POST = ws + OFF_POST;
  float* STAT = ws + OFF_STATS;
  float* Qp   = ws + OFF_Q;
  float* ERAW = ws + OFF_ERAW;
  float* E1   = ws + OFF_E1;
  float* EVEC = ws + OFF_EVEC;
  float* Zp   = ws + OFF_Z;
  float* TSL  = ws + OFF_TSL;
  float* DIST = ws + OFF_DIST;
  int*   IDX0 = (int*)(ws + OFF_IDX0);
  int*   FIDX = (int*)(ws + OFF_FIDX);
  float* SLOT = ws + OFF_SLOTS;
  float* LNB  = ws + OFF_LNB;
  float* QB   = ws + OFF_QB;
  float* UPD  = ws + OFF_UPD;
  float* H1   = ws + OFF_H1;
  float* DPOS = ws + OFF_DPOS;
  float* QP   = ws + OFF_QP;
  float* GI   = ws + OFF_GI;
  float* GH   = ws + OFF_GH;

  float* out = (float*)d_out;
  constexpr long long OUT_SLOTS = (long long)NB * NS * ND * NTOK;   // 67108864
  constexpr long long OUT_IDX   = OUT_SLOTS + (long long)NB * NS * ND;
  constexpr long long OUT_QLOSS = OUT_IDX + NB * NS;

  auto gemm = [&](const float* Aa, const float* Bw, const float* bias, const float* res,
                  float* Cc, int M, int N, int K,
                  long long sA, long long sB, long long sC, int nb, int tB, int relu) {
    dim3 g(N / 128, (M + 127) / 128, nb);
    gemm_wmma_bf16<<<g, 256, 0, stream>>>(Aa, Bw, bias, res, Cc, M, N, K, sA, sB, sC, tB, relu);
  };

  // encoder
  encpos_kernel<<<512, 256, 0, stream>>>(enc_pos_w, enc_pos_b, POS, POST);
  decpos_kernel<<<512, 256, 0, stream>>>(dec_pos_w, dec_pos_b, DPOS);
  stats_kernel<<<NB, 256, 0, stream>>>(x, POST, STAT);
  lnapply_kernel<<<dim3(16, 8, NB), dim3(32, 8), 0, stream>>>(
      x, POS, STAT, enc_norm_g, enc_norm_b, FNP, FB);

  gemm(FNP, wk, bk, nullptr, DV, NB * NTOK, ND, ND, 0, 0, 0, 1, 0, 0);          // k_nopos
  gemm(FB, enc_mlp_w1, enc_mlp_b1, nullptr, KC, NB * NTOK, ND, ND, 0, 0, 0, 1, 0, 1); // t1
  gemm(KC, enc_mlp_w2, enc_mlp_b2, nullptr, FB, NB * NTOK, ND, ND, 0, 0, 0, 1, 0, 0); // fp_mlp
  rowln_kernel<<<NB * NTOK, 256, 0, stream>>>(FB, FB, ni_g, ni_b, ND);           // fp
  gemm(FB, wk, bk, nullptr, KC, NB * NTOK, ND, ND, 0, 0, 0, 1, 0, 0);            // k

  // eigen path
  rownorm_kernel<<<NB * NTOK, 256, 0, stream>>>(DV, ND);                         // xn in-place
  gemm(DV, DV, nullptr, nullptr, COV, NTOK, NTOK, ND,
       (long long)NTOK * ND, (long long)NTOK * ND, (long long)NTOK * NTOK, NB, 1, 0); // cov
  gemm(FB, wv, bv, nullptr, DV, NB * NTOK, ND, ND, 0, 0, 0, 1, 0, 0);            // v (xn dead)

  for (int it = 0; it < 24; it++)
    subspace_kernel<<<NB, 256, 0, stream>>>(COV, Qp, it == 0 ? 1 : 0);
  rayleigh_kernel<<<NB, 256, 0, stream>>>(COV, Qp, ERAW);

  gemm(ERAW, anch_w1, anch_b1, nullptr, E1, NB * NS, NTOK, NTOK, 0, 0, 0, 1, 0, 1);
  gemm(E1, anch_w2, anch_b2, nullptr, EVEC, NB * NS, NTOK, NTOK, 0, 0, 0, 1, 0, 0);
  gemm(EVEC, FNP, nullptr, nullptr, Zp, NS, ND, NTOK,
       (long long)NS * NTOK, (long long)NTOK * ND, (long long)NS * ND, NB, 0, 0);  // z

  // quantization
  gemm(emb, wq, bq, nullptr, TSL, MAXS, ND, ND, 0, 0, 0, 1, 0, 0);               // tslots
  quant_kernel<<<NB * NS, 64, 0, stream>>>(Zp, TSL, DIST, IDX0);
  order_kernel<<<1, 64, 0, stream>>>(DIST, IDX0, FIDX, out + OUT_IDX);
  slotsinit_kernel<<<1024, 256, 0, stream>>>(emb, TSL, Zp, FIDX, SLOT, QP);
  qloss_kernel<<<1, 256, 0, stream>>>(QP, out + OUT_QLOSS);

  // slot attention iterations
  for (int it = 0; it < 3; it++) {
    rowln_kernel<<<NB * NS, 256, 0, stream>>>(SLOT, LNB, ns_g, ns_b, ND);
    gemm(LNB, wq, bq, nullptr, QB, NB * NS, ND, ND, 0, 0, 0, 1, 0, 0);
    attn_kernel<<<NB, 256, 0, stream>>>(QB, KC, DV, UPD);
    gemm(UPD, gru_wih, gru_bih, nullptr, GI, NB * NS, 3 * ND, ND, 0, 0, 0, 1, 1, 0);
    gemm(SLOT, gru_whh, gru_bhh, nullptr, GH, NB * NS, 3 * ND, ND, 0, 0, 0, 1, 1, 0);
    gru_kernel<<<1024, 256, 0, stream>>>(GI, GH, SLOT);
    rowln_kernel<<<NB * NS, 256, 0, stream>>>(SLOT, LNB, npf_g, npf_b, ND);
    gemm(LNB, mlp_w1, mlp_b1, nullptr, H1, NB * NS, ND, ND, 0, 0, 0, 1, 0, 1);
    gemm(H1, mlp_w2, mlp_b2, SLOT, SLOT, NB * NS, ND, ND, 0, 0, 0, 1, 0, 0);     // residual
  }

  // decoder + outputs
  decoder_kernel<<<262144, 256, 0, stream>>>(SLOT, DPOS, out);
  copyslots_kernel<<<1024, 256, 0, stream>>>(SLOT, out + OUT_SLOTS);
}
